// BahdanauAttention_1159641170294
// MI455X (gfx1250) — matrix operand order repaired and checked by reading, not compiled, not run on hardware
//
#include <hip/hip_runtime.h>

#define B_  32
#define S_  2048
#define D_  1024
#define U_  1024

#define MROWS 128                       // s-rows per workgroup (16 per wave)
#define A_BYTES   (MROWS * D_ * 2)      // 262144: A tile, unpadded [128][1024] bf16
#define PANEL_USH (D_ * 16)             // 16384 ushorts = 32 KB per B panel
// LDS total: 262144 + 2*32768 = 327680 = 320 KB exactly

typedef __attribute__((ext_vector_type(8)))  float        v8f;
typedef __attribute__((ext_vector_type(16))) __bf16       v16bf;
typedef __attribute__((ext_vector_type(4)))  unsigned int v4u;

__device__ __forceinline__ unsigned short f2bf(float f) {
  unsigned int u = __float_as_uint(f);
  u += 0x7FFFu + ((u >> 16) & 1u);      // round-to-nearest-even
  return (unsigned short)(u >> 16);
}

// A-matrix gather (16-bit A 16x32 layout): elements 0-7 hold K=base..base+7,
// elements 8-15 hold K=base+16..base+23 -> two 16B chunks 32B apart.
__device__ __forceinline__ v16bf loadA(const unsigned short* p) {
  union { v4u u[2]; v16bf v; } c;
  c.u[0] = *(const v4u*)(p);
  c.u[1] = *(const v4u*)(p + 16);
  return c.v;
}

// B-matrix gather (lane <-> N column, elements <-> K rows base..+7, base+16..+23; row stride 16).
__device__ __forceinline__ v16bf loadB(const unsigned short* p) {
  union { unsigned short s[16]; v16bf v; } c;
#pragma unroll
  for (int i = 0; i < 8; ++i) {
    c.s[i]     = p[16 * i];
    c.s[8 + i] = p[256 + 16 * i];
  }
  return c.v;
}

// ---------------- kernel 1: W1 fp32 -> bf16 (same [D][U] layout) ----------------
__global__ void __launch_bounds__(256)
w1conv_kernel(const float* __restrict__ W1, unsigned short* __restrict__ w1bf) {
  const int i = blockIdx.x * 256 + threadIdx.x;
  w1bf[i] = f2bf(W1[i]);
}

// ---------------- kernel 2: pqv[b][u] = query[b]·W2[:,u] + b1[u] + b2[u] --------
__global__ void __launch_bounds__(256)
projq_kernel(const float* __restrict__ q, const float* __restrict__ W2,
             const float* __restrict__ b1, const float* __restrict__ b2,
             float* __restrict__ pqv) {
  const int u = blockIdx.x * 256 + threadIdx.x;
  const int b = blockIdx.y;
  const float* qr = q + (size_t)b * D_;
  float acc = b1[u] + b2[u];
#pragma unroll 4
  for (int d = 0; d < D_; ++d)
    acc = fmaf(qr[d], W2[(size_t)d * U_ + u], acc);
  pqv[(size_t)b * U_ + u] = acc;
}

// ---------------- kernel 3: scores[b][s] via WMMA bf16 --------------------------
// WG = (batch, 128 s-rows), 8 waves x 16 rows. A lives in 256 VGPRs per wave.
// B panels (16 u x 1024 k) double-buffered in LDS via register staging, one
// barrier per panel. B tiles consumed in pairs with a depth-2 pair pipeline
// (4 tiles live -> distinct VGPRs, dscnt waits > 0).
__global__ void __launch_bounds__(256, 2)
attn_scores_kernel(const float* __restrict__ values,
                   const unsigned short* __restrict__ w1bf,
                   const float* __restrict__ pqv,
                   const float* __restrict__ Vw,
                   float* __restrict__ scores) {
  extern __shared__ char smem[];
  unsigned short* A  = (unsigned short*)smem;               // [128][1024] bf16
  unsigned short* Bt = (unsigned short*)(smem + A_BYTES);   // 2 panels [1024][16]

  const int b   = blockIdx.y;
  const int s0  = blockIdx.x * MROWS;
  const int tid = threadIdx.x;

  // ---- Stage values[b, s0:s0+128, :] -> LDS bf16; 8 float4/thread/iter for MLP.
  const float4* src = (const float4*)(values + ((size_t)b * S_ + s0) * D_);
#pragma unroll 1
  for (int it = 0; it < 16; ++it) {
    const int i0 = it * 2048 + tid * 8;       // float4 index; stays in one row
    float4 vv[8];
#pragma unroll
    for (int q = 0; q < 8; ++q) vv[q] = src[i0 + q];
    unsigned pk[16];
#pragma unroll
    for (int q = 0; q < 8; ++q) {
      pk[2 * q]     = (unsigned)f2bf(vv[q].x) | ((unsigned)f2bf(vv[q].y) << 16);
      pk[2 * q + 1] = (unsigned)f2bf(vv[q].z) | ((unsigned)f2bf(vv[q].w) << 16);
    }
    v4u* dst = (v4u*)(A + (size_t)i0 * 4);    // unpadded A => linear mapping
#pragma unroll
    for (int q = 0; q < 4; ++q) dst[q] = ((const v4u*)pk)[q];
  }

  // ---- Panel 0 into registers (independent of LDS; overlaps A staging).
  v4u treg[8];
#pragma unroll
  for (int i = 0; i < 8; ++i) {
    const int c = tid + 256 * i;              // chunk: k = c>>1, half = c&1
    treg[i] = *(const v4u*)(w1bf + (size_t)(c >> 1) * U_ + 8 * (c & 1));
  }

  __syncthreads();                            // A visible

  const int lane = tid & 31;
  const int wave = tid >> 5;                  // owns s rows 16*wave .. +15
  const int n    = lane & 15;
  const int h    = lane >> 4;

  // ---- One-time A preload into 256 VGPRs (reused across all 64 panels).
  const unsigned short* arow = A + (size_t)(16 * wave + n) * D_ + 8 * h;
  v16bf areg[32];
#pragma unroll
  for (int k = 0; k < 32; ++k) areg[k] = loadA(arow + 32 * k);

  // ---- Store panel 0 into buffer 0.
  {
    v4u* d0 = (v4u*)Bt;
#pragma unroll
    for (int i = 0; i < 8; ++i) d0[tid + 256 * i] = treg[i];
  }
  __syncthreads();                            // panel 0 visible

  float sacc[8] = {0.f, 0.f, 0.f, 0.f, 0.f, 0.f, 0.f, 0.f};

#pragma unroll 1
  for (int j = 0; j < 64; ++j) {
    // Load panel j+1 into regs (clamped at the tail: scalar select, no branch,
    // keeps the whole loop body one scheduling region for the sched template).
    const int jn  = (j < 63) ? (j + 1) : j;
    const int u0n = jn * 16;
#pragma unroll
    for (int i = 0; i < 8; ++i) {
      const int c = tid + 256 * i;
      const unsigned short* g = w1bf + (size_t)(c >> 1) * U_ + u0n + 8 * (c & 1);
      treg[i] = *(const v4u*)g;
      __builtin_prefetch(g + 16, 0, 3);       // panel j+2 -> caches
    }

    // Compute on current panel. Consume B tiles in pairs: both tiles of a pair
    // are live before the first WMMA of the pair -> RA must use distinct VGPRs.
    const unsigned short* bp = Bt + (size_t)(j & 1) * PANEL_USH + n + 128 * h;
    v8f cacc = {};
#pragma unroll
    for (int k = 0; k < 32; k += 2) {
      v16bf bv0 = loadB(bp + 512 * k);        // rows 32k + 8h.. of panel, column n
      v16bf bv1 = loadB(bp + 512 * (k + 1));
      cacc = __builtin_amdgcn_wmma_f32_16x16x32_bf16(false, areg[k], false, bv0,
                                                     (short)0, cacc, false, false);
      cacc = __builtin_amdgcn_wmma_f32_16x16x32_bf16(false, areg[k + 1], false, bv1,
                                                     (short)0, cacc, false, false);
    }

    // Dump staged panel into the other buffer (dead write on the last iter).
    {
      v4u* dn = (v4u*)(Bt + (size_t)(jn & 1) * PANEL_USH);
#pragma unroll
      for (int i = 0; i < 8; ++i) dn[tid + 256 * i] = treg[i];
    }

    // Fused epilogue: tanh(pv + pq) * V[u], accumulate per s-row.
    const int u0 = j * 16;
    const float pq = pqv[(size_t)b * U_ + u0 + n];
    const float vv = Vw[u0 + n];
#pragma unroll
    for (int r = 0; r < 8; ++r)
      sacc[r] += tanhf(cacc[r] + pq) * vv;

    // Scheduling template: panel global loads first; preload 2 tile-pairs
    // (8 DS reads), then [2 WMMA][4 DS reads for pair+2] x14, tail WMMAs,
    // then the panel DS writes.
    __builtin_amdgcn_sched_group_barrier(0x020, 8, 0);   // 8 VMEM reads (panel)
    __builtin_amdgcn_sched_group_barrier(0x100, 8, 0);   // DS reads: pairs 0,1
#pragma unroll
    for (int g = 0; g < 14; ++g) {
      __builtin_amdgcn_sched_group_barrier(0x008, 2, 0); // WMMA pair g
      __builtin_amdgcn_sched_group_barrier(0x100, 4, 0); // DS reads: pair g+2
    }
    __builtin_amdgcn_sched_group_barrier(0x008, 4, 0);   // WMMA pairs 14,15
    __builtin_amdgcn_sched_group_barrier(0x200, 4, 0);   // DS writes (panel)

    __syncthreads();                          // panel consumed / next visible
  }

  // Reduce over N (16 lanes per half-wave); rows are wave-private -> plain store.
#pragma unroll
  for (int r = 0; r < 8; ++r) {
    float t = sacc[r];
    t += __shfl_xor(t, 1, 32);
    t += __shfl_xor(t, 2, 32);
    t += __shfl_xor(t, 4, 32);
    t += __shfl_xor(t, 8, 32);
    if (n == 0)
      scores[(size_t)b * S_ + s0 + 16 * wave + 8 * h + r] = t;
  }
}

// ---------------- kernel 4: softmax over S per batch; zero context region -------
__global__ void __launch_bounds__(256)
softmax_kernel(const float* __restrict__ scores, float* __restrict__ out) {
  const int b   = blockIdx.x;
  const int tid = threadIdx.x;
  __shared__ float red[256];
  const float* sc = scores + (size_t)b * S_;

  float v[8];
  float mx = -3.4e38f;
#pragma unroll
  for (int i = 0; i < 8; ++i) { v[i] = sc[tid + 256 * i]; mx = fmaxf(mx, v[i]); }
  red[tid] = mx; __syncthreads();
  for (int s = 128; s > 0; s >>= 1) {
    if (tid < s) red[tid] = fmaxf(red[tid], red[tid + s]);
    __syncthreads();
  }
  mx = red[0]; __syncthreads();

  float sum = 0.f;
#pragma unroll
  for (int i = 0; i < 8; ++i) { v[i] = __expf(v[i] - mx); sum += v[i]; }
  red[tid] = sum; __syncthreads();
  for (int s = 128; s > 0; s >>= 1) {
    if (tid < s) red[tid] += red[tid + s];
    __syncthreads();
  }
  const float inv = 1.f / red[0];

  float* attn = out + (size_t)B_ * D_ + (size_t)b * S_;
#pragma unroll
  for (int i = 0; i < 8; ++i) attn[tid + 256 * i] = v[i] * inv;

  // zero context accumulator region for kernel 5
  for (int i = tid; i < D_; i += 256) out[(size_t)b * D_ + i] = 0.f;
}

// ---------------- kernel 5: context[b][d] = sum_s attn[b][s] * values[b][s][d] --
__global__ void __launch_bounds__(256)
context_kernel(const float* __restrict__ values, float* __restrict__ out) {
  const int d  = blockIdx.x * 256 + threadIdx.x;
  const int b  = blockIdx.y;
  const int s0 = blockIdx.z * 256;
  const float* attn = out + (size_t)B_ * D_ + (size_t)b * S_ + s0;
  const float* vp   = values + ((size_t)b * S_ + s0) * D_ + d;
  float acc = 0.f;
#pragma unroll 4
  for (int s = 0; s < 256; ++s)
    acc += attn[s] * vp[(size_t)s * D_];
  atomicAdd(&out[(size_t)b * D_ + d], acc);
}

// --------------------------------------------------------------------------------
extern "C" void kernel_launch(void* const* d_in, const int* in_sizes, int n_in,
                              void* d_out, int out_size, void* d_ws, size_t ws_size,
                              hipStream_t stream) {
  (void)in_sizes; (void)n_in; (void)out_size; (void)ws_size;
  const float* query  = (const float*)d_in[0];
  const float* values = (const float*)d_in[1];
  const float* W1     = (const float*)d_in[2];
  const float* b1     = (const float*)d_in[3];
  const float* W2     = (const float*)d_in[4];
  const float* b2     = (const float*)d_in[5];
  const float* Vw     = (const float*)d_in[6];
  // d_in[7] = bV: softmax is shift-invariant and raw scores are not outputs.

  float* out = (float*)d_out;
  char*  ws  = (char*)d_ws;
  unsigned short* w1bf = (unsigned short*)ws;                                // 2 MB
  float* pqv    = (float*)(ws + (size_t)D_ * U_ * 2);                        // 128 KB
  float* scores = (float*)(ws + (size_t)D_ * U_ * 2 + (size_t)B_ * U_ * 4);  // 256 KB

  w1conv_kernel<<<(D_ * U_) / 256, 256, 0, stream>>>(W1, w1bf);
  projq_kernel<<<dim3(U_ / 256, B_), 256, 0, stream>>>(query, W2, b1, b2, pqv);

  const size_t shmem = (size_t)A_BYTES + 2 * PANEL_USH * 2;  // 327680 = 320 KB
  attn_scores_kernel<<<dim3(S_ / MROWS, B_), 256, shmem, stream>>>(
      values, w1bf, pqv, Vw, scores);

  softmax_kernel<<<B_, 256, 0, stream>>>(scores, out);
  context_kernel<<<dim3(D_ / 256, B_, S_ / 256), 256, 0, stream>>>(values, out);
}
